// LFWLWrapper_19679540150600
// MI455X (gfx1250) — matrix operand order repaired and checked by reading, not compiled, not run on hardware
//
#include <hip/hip_runtime.h>
#include <stdint.h>

// ---- problem constants (match reference) ----
#define Bq 32
#define Nq 96
#define Dq 64
#define Lq 5
#define PAIRS (Nq * Nq)            // 9216 pairs per graph
#define RTOT (Bq * PAIRS)          // 294912 pair-rows total
#define ELEMS ((long)RTOT * Dq)    // 18,874,368 elements of W

typedef __bf16 bf16;
typedef __attribute__((ext_vector_type(16))) __bf16 v16bf;
typedef __attribute__((ext_vector_type(8)))  float  v8f;

// Native f32 -> bf16 (fptrunc, RNE) — lowers to the hardware convert on gfx1250
// instead of a 4-op manual round-to-nearest-even sequence.
__device__ __forceinline__ bf16 f2bf(float f) { return (bf16)f; }

// A-fragment (16x32 bf16, row-major source, row stride ld elements).
// Lane L<16: M=L, K=0..15; lane L>=16: M=L-16, K=16..31.
__device__ __forceinline__ v16bf load_frag_a(const bf16* __restrict__ p, int ld) {
  int lane = threadIdx.x & 31;
  int r  = lane & 15;
  int kh = lane >> 4;
  const bf16* row = p + (long)r * ld + kh * 16;
  union { uint4 q; bf16 h[8]; } lo, hi;
  lo.q = *reinterpret_cast<const uint4*>(row);        // 16B aligned
  hi.q = *reinterpret_cast<const uint4*>(row + 8);
  v16bf a;
#pragma unroll
  for (int i = 0; i < 8; ++i) { a[i] = lo.h[i]; a[i + 8] = hi.h[i]; }
  return a;
}

// B-fragment from a row-major N x K (i.e. B-transposed) source, row stride ld.
// Lane L<16: N=L, K=0..15; lane L>=16: N=L-16, K=16..31 -> 32 contiguous bytes.
__device__ __forceinline__ v16bf load_frag_bt(const bf16* p, int ld) {
  int lane = threadIdx.x & 31;
  int n  = lane & 15;
  int k0 = (lane >> 4) * 16;
  const bf16* row = p + (long)n * ld + k0;
  union { uint4 q[2]; bf16 h[16]; } u;
  u.q[0] = *reinterpret_cast<const uint4*>(row);
  u.q[1] = *reinterpret_cast<const uint4*>(row + 8);
  v16bf b;
#pragma unroll
  for (int i = 0; i < 16; ++i) b[i] = u.h[i];
  return b;
}

__device__ __forceinline__ v8f wmma_bf16(v16bf a, v16bf b, v8f c) {
  return __builtin_amdgcn_wmma_f32_16x16x32_bf16(false, a, false, b, (short)0, c,
                                                 false, false);
}

// Stage a [64(k),64(e)] fp32 weight into LDS in WMMA B-fragment order:
// wlf[((ct*2+ks)*32 + lane)*16 + i] = bf16(wgt[(ks*32 + (lane>>4)*16 + i)*64 + ct*16 + (lane&15)])
__device__ __forceinline__ void stage_weights(const float* __restrict__ wgt,
                                              bf16* __restrict__ wlf) {
  for (int idx = threadIdx.x; idx < Dq * Dq; idx += 256) {
    int i  = idx & 15;
    int ln = (idx >> 4) & 31;
    int ks = (idx >> 9) & 1;
    int ct = idx >> 10;
    int k = ks * 32 + ((ln >> 4) * 16) + i;
    int e = ct * 16 + (ln & 15);
    wlf[idx] = f2bf(wgt[k * Dq + e]);
  }
}

// ---------------- dense adjacency build ----------------
__global__ void k_fill_labels(int* __restrict__ A) {
  int i = blockIdx.x * 256 + threadIdx.x;
  if (i < RTOT) A[i] = 5;  // EDGE_VOCAB = no-edge
}

__global__ void k_scatter_edges(const int* __restrict__ ei, const int* __restrict__ ea,
                                int* __restrict__ A, int E) {
  int e = blockIdx.x * 256 + threadIdx.x;
  if (e >= E) return;
  int s = ei[e], t = ei[E + e];
  int g = s / Nq, u = s % Nq, v = t % Nq;
  int lab = ea[e];
  A[(g * Nq + u) * Nq + v] = lab;
  A[(g * Nq + v) * Nq + u] = lab;   // symmetrize
}

// ---------------- pair-state init: W = h_u + h_v + edge_emb[A] ----------------
__global__ void k_init_W(const int* __restrict__ x, const float* __restrict__ nemb,
                         const float* __restrict__ eemb, const int* __restrict__ A,
                         float* __restrict__ W, bf16* __restrict__ Wbf) {
  long idx = (long)blockIdx.x * 256 + threadIdx.x;
  int e = (int)(idx & 63);
  long pair = idx >> 6;
  int v = (int)(pair % Nq);
  long t = pair / Nq;
  int u = (int)(t % Nq);
  int b = (int)(t / Nq);
  int xu = x[b * Nq + u], xv = x[b * Nq + v];
  int lab = A[pair];
  float val = nemb[xu * Dq + e] + nemb[xv * Dq + e] + eemb[lab * Dq + e];
  W[idx] = val;
  Wbf[idx] = f2bf(val);
}

// ---------------- channel GEMM -> relu -> d-major scatter (h1 / h2) -------------
// H2 == false: out[((b*64+e)*96 + i)*96 + j]   (h1t, [b,d,u,w], b128 stores)
// H2 == true : out[((b*64+e)*96 + j)*96 + i]*adj (h2tT, [b,d,v,w], strided stores)
template <bool H2>
__global__ void k_rowgemm_scatter(const bf16* __restrict__ in,   // [RTOT,64]
                                  const float* __restrict__ wgt, // [64,64] (d,e)
                                  const float* __restrict__ bias,// [64]
                                  const int*  __restrict__ A,    // [RTOT] labels
                                  bf16* __restrict__ outT) {
  __shared__ __align__(32) bf16 wlf[Dq * Dq];
  stage_weights(wgt, wlf);
  __syncthreads();

  int wave = threadIdx.x >> 5;
  int lane = threadIdx.x & 31;
  long r0 = ((long)blockIdx.x * 8 + wave) * 16;      // 16 consecutive pair-rows
  v16bf a0 = load_frag_a(in + r0 * Dq, Dq);          // K = 0..31
  v16bf a1 = load_frag_a(in + r0 * Dq + 32, Dq);     // K = 32..63

  int b    = (int)(r0 / PAIRS);
  int rem  = (int)(r0 % PAIRS);
  int irow = rem / Nq;
  int j0   = rem % Nq;                               // multiple of 16
  int cc = lane & 15;
  int m0 = (lane >> 4) * 8;

  float adjmask[8];
  if (H2) {
#pragma unroll
    for (int i = 0; i < 8; ++i)
      adjmask[i] = (A[r0 + m0 + i] != 5) ? 1.0f : 0.0f;
  }

  const v16bf* wfrag = reinterpret_cast<const v16bf*>(wlf);
#pragma unroll
  for (int ct = 0; ct < 4; ++ct) {
    v16bf b0 = wfrag[(ct * 2 + 0) * 32 + lane];      // vector ds loads
    v16bf b1 = wfrag[(ct * 2 + 1) * 32 + lane];
    v8f c = {};
    c = wmma_bf16(a0, b0, c);
    c = wmma_bf16(a1, b1, c);

    int e = ct * 16 + cc;
    float be = bias[e];
    if (!H2) {
      union { uint4 q; bf16 h[8]; } st;
#pragma unroll
      for (int i = 0; i < 8; ++i)
        st.h[i] = f2bf(fmaxf(c[i] + be, 0.0f));
      long ob = (((long)b * Dq + e) * Nq + irow) * Nq + j0 + m0;
      *reinterpret_cast<uint4*>(outT + ob) = st.q;   // one b128 store
    } else {
      long ob = (((long)b * Dq + e) * Nq + (j0 + m0)) * Nq + irow;
#pragma unroll
      for (int i = 0; i < 8; ++i) {
        float v = fmaxf(c[i] + be, 0.0f) * adjmask[i];
        outT[ob + (long)i * Nq] = f2bf(v);           // transposed scatter
      }
    }
  }
}

// ------- local FWL2 contraction: 2048 x (96x96x96) bf16 GEMMs ----------------
// block = one (b,d) batch, 6 waves; wave = u-tile, sweeps 6 v-tiles.
// The 18KB B panel is staged once into LDS (6x reuse across waves).
__global__ void k_contract(const bf16* __restrict__ h1t,   // [B*64, 96(u), 96(w)]
                           const bf16* __restrict__ h2tT,  // [B*64, 96(v), 96(w)]
                           bf16* __restrict__ Mout) {      // [B,96,96,64]
  __shared__ __align__(16) bf16 Bs[PAIRS];                 // 18KB (of 320KB WGP LDS)
  int ut   = threadIdx.x >> 5;       // 0..5
  int lane = threadIdx.x & 31;
  long bd  = blockIdx.x;             // (b,d)

  // cooperative stage of the whole 96x96 B panel: 1152 x 16B chunks / 192 threads
  const bf16* Bg = h2tT + bd * PAIRS;
#pragma unroll
  for (int c = threadIdx.x; c < PAIRS / 8; c += 192)
    *reinterpret_cast<uint4*>(&Bs[c * 8]) =
        *reinterpret_cast<const uint4*>(Bg + c * 8);
  // warm L2 for the next block's panel while we compute
  if (bd + 1 < (long)Bq * Dq)
    __builtin_prefetch(Bg + PAIRS, 0, 1);                  // global_prefetch_b8

  const bf16* Ab = h1t + bd * PAIRS + (long)ut * 16 * Nq;
  v16bf a0 = load_frag_a(Ab,      Nq);   // K = w 0..31
  v16bf a1 = load_frag_a(Ab + 32, Nq);   // K = w 32..63
  v16bf a2 = load_frag_a(Ab + 64, Nq);   // K = w 64..95
  __syncthreads();

  int d  = (int)(bd & 63);
  int b_ = (int)(bd >> 6);
  int vv = lane & 15;
  int m0 = (lane >> 4) * 8;

#pragma unroll
  for (int vt = 0; vt < 6; ++vt) {
    const bf16* Bb = &Bs[vt * 16 * Nq];
    v8f c = {};
    c = wmma_bf16(a0, load_frag_bt(Bb,      Nq), c);   // B frags via ds_load_b128
    c = wmma_bf16(a1, load_frag_bt(Bb + 32, Nq), c);
    c = wmma_bf16(a2, load_frag_bt(Bb + 64, Nq), c);
    int v = vt * 16 + vv;
#pragma unroll
    for (int i = 0; i < 8; ++i) {
      int u = ut * 16 + m0 + i;
      Mout[((((long)b_ * Nq + u) * Nq + v) << 6) + d] = f2bf(c[i]);
    }
  }
}

// ---------------- W += M @ Wm + bm ----------------
__global__ void k_rowgemm_update(const bf16* __restrict__ in,   // M [RTOT,64]
                                 const float* __restrict__ wgt, // Wm [64,64]
                                 const float* __restrict__ bias,
                                 float* __restrict__ W) {
  __shared__ __align__(32) bf16 wlf[Dq * Dq];
  stage_weights(wgt, wlf);
  __syncthreads();
  int wave = threadIdx.x >> 5;
  int lane = threadIdx.x & 31;
  long r0 = ((long)blockIdx.x * 8 + wave) * 16;
  v16bf a0 = load_frag_a(in + r0 * Dq, Dq);
  v16bf a1 = load_frag_a(in + r0 * Dq + 32, Dq);
  int cc = lane & 15;
  int m0 = (lane >> 4) * 8;
  const v16bf* wfrag = reinterpret_cast<const v16bf*>(wlf);
#pragma unroll
  for (int ct = 0; ct < 4; ++ct) {
    v16bf b0 = wfrag[(ct * 2 + 0) * 32 + lane];
    v16bf b1 = wfrag[(ct * 2 + 1) * 32 + lane];
    v8f c = {};
    c = wmma_bf16(a0, b0, c);
    c = wmma_bf16(a1, b1, c);
    int e = ct * 16 + cc;
    float be = bias[e];
#pragma unroll
    for (int i = 0; i < 8; ++i) {
      long idx = (r0 + m0 + i) * Dq + e;
      W[idx] = W[idx] + c[i] + be;
    }
  }
}

// ------- instance-norm stats: coalesced, thread t pinned to channel t&63 -------
__global__ void k_reduce_stats(const float* __restrict__ W, float* __restrict__ stats) {
  int b = blockIdx.x;
  int t = threadIdx.x;
  const float* base = W + (long)b * PAIRS * Dq;
  float s = 0.0f, sq = 0.0f;
  for (int k = 0; k < PAIRS * Dq; k += 256) {   // 256 % 64 == 0 -> channel fixed
    float x = base[k + t];
    s += x; sq += x * x;
  }
  __shared__ float ls[256], lq[256];
  ls[t] = s; lq[t] = sq;
  __syncthreads();
  if (t < Dq) {
    float S = ls[t] + ls[t + 64] + ls[t + 128] + ls[t + 192];
    float Q = lq[t] + lq[t + 64] + lq[t + 128] + lq[t + 192];
    float mu  = S * (1.0f / PAIRS);
    float var = Q * (1.0f / PAIRS) - mu * mu;
    stats[(b * Dq + t) * 2]     = mu;
    stats[(b * Dq + t) * 2 + 1] = rsqrtf(var + 1e-5f);
  }
}

__global__ void k_norm_relu(float* __restrict__ W, bf16* __restrict__ Wbf,
                            const float* __restrict__ stats,
                            const float* __restrict__ g, const float* __restrict__ bt) {
  long idx = (long)blockIdx.x * 256 + threadIdx.x;
  int e = (int)(idx & 63);
  long pair = idx >> 6;
  int b = (int)(pair / PAIRS);
  const float* st = stats + ((long)b * Dq + e) * 2;
  float x = (W[idx] - st[0]) * st[1] * g[e] + bt[e];
  x = fmaxf(x, 0.0f);
  W[idx] = x;
  Wbf[idx] = f2bf(x);
}

// ---------------- pooling: sum over pairs, Linear(D,1) ----------------
__global__ void k_pool(const float* __restrict__ W, const float* __restrict__ pw,
                       const float* __restrict__ pb, float* __restrict__ out) {
  __shared__ float spw[Dq];
  __shared__ float red[256];
  if (threadIdx.x < Dq) spw[threadIdx.x] = pw[threadIdx.x];
  __syncthreads();
  const float* base = W + (long)blockIdx.x * PAIRS * Dq;
  float acc = 0.0f;
  for (long i = threadIdx.x; i < (long)PAIRS * Dq; i += 256)
    acc += base[i] * spw[(int)(i & 63)];
  red[threadIdx.x] = acc;
  __syncthreads();
  for (int off = 128; off > 0; off >>= 1) {
    if (threadIdx.x < off) red[threadIdx.x] += red[threadIdx.x + off];
    __syncthreads();
  }
  if (threadIdx.x == 0) out[blockIdx.x] = red[0] + pb[0];
}

extern "C" void kernel_launch(void* const* d_in, const int* in_sizes, int n_in,
                              void* d_out, int out_size, void* d_ws, size_t ws_size,
                              hipStream_t stream) {
  const int*   x        = (const int*)  d_in[0];
  const int*   ei       = (const int*)  d_in[1];
  const int*   ea       = (const int*)  d_in[2];
  const float* node_emb = (const float*)d_in[3];
  const float* edge_emb = (const float*)d_in[4];
  const float* W1 = (const float*)d_in[5];
  const float* b1 = (const float*)d_in[6];
  const float* W2 = (const float*)d_in[7];
  const float* b2 = (const float*)d_in[8];
  const float* Wm = (const float*)d_in[9];
  const float* bm = (const float*)d_in[10];
  const float* gamma = (const float*)d_in[11];
  const float* beta  = (const float*)d_in[12];
  const float* pool_w = (const float*)d_in[13];
  const float* pool_b = (const float*)d_in[14];
  float* out = (float*)d_out;

  // workspace carve-out (~190 MB total)
  char* p = (char*)d_ws;
  auto alloc = [&](size_t bytes) {
    char* q = p;
    p += (bytes + 255) & ~(size_t)255;
    return q;
  };
  int*   A     = (int*)  alloc((size_t)RTOT * sizeof(int));      //  1.2 MB
  float* W     = (float*)alloc((size_t)ELEMS * sizeof(float));   // 75.5 MB
  bf16*  Wbf   = (bf16*) alloc((size_t)ELEMS * sizeof(bf16));    // 37.7 MB (doubles as M buffer)
  bf16*  h1t   = (bf16*) alloc((size_t)ELEMS * sizeof(bf16));    // 37.7 MB
  bf16*  h2tT  = (bf16*) alloc((size_t)ELEMS * sizeof(bf16));    // 37.7 MB
  float* stats = (float*)alloc((size_t)Bq * Dq * 2 * sizeof(float));
  bf16*  Mbf   = Wbf;  // Wbf is dead between the h1/h2 GEMMs and the norm rewrite

  const int E = in_sizes[2];                 // B*EP directed edges
  const int elemBlocks = (int)(ELEMS / 256);           // 73728
  const int gemmBlocks = RTOT / 16 / 8;                // 2304 (8 waves/block, 16 rows/wave)

  k_fill_labels<<<(RTOT + 255) / 256, 256, 0, stream>>>(A);
  k_scatter_edges<<<(E + 255) / 256, 256, 0, stream>>>(ei, ea, A, E);
  k_init_W<<<elemBlocks, 256, 0, stream>>>(x, node_emb, edge_emb, A, W, Wbf);

  for (int l = 0; l < Lq; ++l) {
    k_rowgemm_scatter<false><<<gemmBlocks, 256, 0, stream>>>(
        Wbf, W1 + (size_t)l * Dq * Dq, b1 + (size_t)l * Dq, nullptr, h1t);
    k_rowgemm_scatter<true><<<gemmBlocks, 256, 0, stream>>>(
        Wbf, W2 + (size_t)l * Dq * Dq, b2 + (size_t)l * Dq, A, h2tT);
    k_contract<<<Bq * Dq, 192, 0, stream>>>(h1t, h2tT, Mbf);   // 6 waves = 6 u-tiles
    k_rowgemm_update<<<gemmBlocks, 256, 0, stream>>>(
        Mbf, Wm + (size_t)l * Dq * Dq, bm + (size_t)l * Dq, W);
    k_reduce_stats<<<Bq, 256, 0, stream>>>(W, stats);
    k_norm_relu<<<elemBlocks, 256, 0, stream>>>(
        W, Wbf, stats, gamma + (size_t)l * Dq, beta + (size_t)l * Dq);
  }

  k_pool<<<Bq, 256, 0, stream>>>(W, pool_w, pool_b, out);
}
